// NodeModel_63170378990109
// MI455X (gfx1250) — matrix operand order repaired and checked by reading, not compiled, not run on hardware
//
#include <hip/hip_runtime.h>
#include <hip/hip_bf16.h>

typedef _Float16 v16h __attribute__((ext_vector_type(16)));
typedef _Float16 h8   __attribute__((ext_vector_type(8)));
typedef float    v8f  __attribute__((ext_vector_type(8)));
typedef float    f4   __attribute__((ext_vector_type(4)));
typedef int      v4i  __attribute__((ext_vector_type(4)));

#define F_IN        64
#define K1          128           // 2*F_IN
#define HIDDEN      128
#define F_OUT       128
#define E_PER_BLOCK 128
#define THREADS     256
#define LSTR        136           // padded LDS row stride in halves (272B, 16B-aligned)

// ---------------------------------------------------------------------------
// CDNA5 async global->LDS copy (GLOBAL_LOAD_ASYNC_TO_LDS_B128, ASYNCcnt),
// with a b128 VGPR-bounce fallback if the builtin is absent on this toolchain.
// Builtin prototype (from hipcc diagnostic): (v4i AS1*, v4i AS3*, Ii, Ii).
// ---------------------------------------------------------------------------
#if __has_builtin(__builtin_amdgcn_global_load_async_to_lds_b128)
#define HAS_ASYNC_LDS 1
#else
#define HAS_ASYNC_LDS 0
#endif

__device__ __forceinline__ void copy16_g2lds(const _Float16* g, _Float16* l) {
#if HAS_ASYNC_LDS
    __builtin_amdgcn_global_load_async_to_lds_b128(
        (__attribute__((address_space(1))) v4i*)g,
        (__attribute__((address_space(3))) v4i*)l, 0, 0);
#else
    *(h8*)l = *(const h8*)g;
#endif
}

__device__ __forceinline__ void wait_async_copies() {
#if HAS_ASYNC_LDS
#if __has_builtin(__builtin_amdgcn_s_wait_asynccnt)
    __builtin_amdgcn_s_wait_asynccnt(0);
#else
    asm volatile("s_wait_asynccnt 0" ::: "memory");
#endif
#endif
}

// ---------------------------------------------------------------------------
// Prep: W[k][n] (fp32, row-major [in,out]) -> Wt[n][k] (f16) in workspace.
// Run once per launch; 32 KB per weight, trivially cheap.
// ---------------------------------------------------------------------------
__global__ __launch_bounds__(THREADS) void prep_weights(
    const float* __restrict__ W1, const float* __restrict__ W2,
    _Float16* __restrict__ w1t, _Float16* __restrict__ w2t)
{
    int idx = blockIdx.x * THREADS + threadIdx.x;   // 0 .. 32767
    int which = idx >> 14;
    int j = idx & 16383;                            // 0 .. 16383
    int k = j >> 7;                                 // input dim (row of W)
    int n = j & 127;                                // output dim (col of W)
    if (which == 0) w1t[n * K1 + k]     = (_Float16)W1[j];
    else            w2t[n * HIDDEN + k] = (_Float16)W2[j];
}

// ---------------------------------------------------------------------------
// Main fused kernel: gather+concat -> GEMM1+bias+ReLU -> GEMM2+bias -> out
// ---------------------------------------------------------------------------
__global__ __launch_bounds__(THREADS, 2) void edge_mlp(
    const float* __restrict__ x,        // [N_NODES, 64]
    const int*   __restrict__ erow,     // edge_index row 0, [E]
    const float* __restrict__ ea,       // [E, 64]
    const float* __restrict__ b1,       // [128]
    const float* __restrict__ b2,       // [128]
    const _Float16* __restrict__ w1t,   // [128][128] f16, [n][k]
    const _Float16* __restrict__ w2t,   // [128][128] f16, [n][k]
    float* __restrict__ out,            // [E, 128]
    int nE)
{
    // feats tile (later reused as hidden tile), padded rows
    __shared__ __align__(16) _Float16 sA [E_PER_BLOCK * LSTR]; // 34816 B
    __shared__ __align__(16) _Float16 sW1[HIDDEN * LSTR];      // 34816 B
    __shared__ __align__(16) _Float16 sW2[F_OUT  * LSTR];      // 34816 B
    __shared__ float sB1[HIDDEN];
    __shared__ float sB2[F_OUT];

    const int tid = threadIdx.x;
    const int e0  = blockIdx.x * E_PER_BLOCK;

    // ---- weights: async DMA straight into padded LDS tiles ----
    // 2048 16B-chunks per weight; 8 per thread per weight.
    #pragma unroll
    for (int i = tid; i < 2048; i += THREADS) {
        int n  = i >> 4;          // LDS row (output dim)
        int kc = i & 15;          // 16B chunk within row
        copy16_g2lds(w1t + i * 8, sW1 + n * LSTR + kc * 8);
        copy16_g2lds(w2t + i * 8, sW2 + n * LSTR + kc * 8);
    }
    if (tid < HIDDEN) sB1[tid] = b1[tid];
    if (tid < F_OUT)  sB2[tid] = b2[tid];

    // ---- gather + concat + fp32->f16 into sA: 2 threads per edge ----
    {
        const int eloc = tid >> 1;           // 0..127
        const int half = tid & 1;            // 0: x[row], 1: edge_attr
        const int e    = e0 + eloc;
        _Float16* dst = sA + eloc * LSTR + half * F_IN;
        if (e < nE) {
            const f4* s4 = half ? (const f4*)(ea + (size_t)e * F_IN)
                                : (const f4*)(x  + (size_t)erow[e] * F_IN);
            #pragma unroll
            for (int j = 0; j < 8; ++j) {
                f4 va, vb;
                if (half) {  // edge_attr is streamed exactly once: non-temporal
                    va = __builtin_nontemporal_load(s4 + 2 * j);
                    vb = __builtin_nontemporal_load(s4 + 2 * j + 1);
                } else {     // x table is 12.8 MB: keep it L2-resident
                    va = s4[2 * j];
                    vb = s4[2 * j + 1];
                }
                h8 h = { (_Float16)va.x, (_Float16)va.y, (_Float16)va.z, (_Float16)va.w,
                         (_Float16)vb.x, (_Float16)vb.y, (_Float16)vb.z, (_Float16)vb.w };
                *(h8*)(dst + 8 * j) = h;
            }
        } else {
            #pragma unroll
            for (int j = 0; j < 8; ++j) *(h8*)(dst + 8 * j) = (h8)0;
        }
    }
    wait_async_copies();      // ASYNCcnt==0: weight tiles resident in LDS
    __syncthreads();

    // ---- per-wave tiling: wave w owns edge rows [16w, 16w+16) ----
    const int lane  = tid & 31;
    const int wv    = tid >> 5;          // M-tile index, 0..7
    const int lhalf = lane >> 4;         // 0 | 1
    const int l15   = lane & 15;
    const int akoff = lhalf ? 8  : 0;    // A: K chunks {k0..k0+7, k0+16..k0+23}
    const int bkoff = lhalf ? 8  : 0;    // B: K chunks {k0..k0+7, k0+16..k0+23}

    const _Float16* aRow = sA + (wv * 16 + l15) * LSTR;

    auto loadA = [&](int kbase) {
        h8 lo = *(const h8*)(aRow + kbase + akoff);
        h8 hi = *(const h8*)(aRow + kbase + akoff + 16);
        v16h r;
        #pragma unroll
        for (int i = 0; i < 8; ++i) { r[i] = lo[i]; r[8 + i] = hi[i]; }
        return r;
    };
    auto loadB = [&](const _Float16* wt, int nt, int kbase) {
        const _Float16* p = wt + (nt * 16 + l15) * LSTR + kbase + bkoff;
        h8 lo = *(const h8*)(p);
        h8 hi = *(const h8*)(p + 16);
        v16h r;
        #pragma unroll
        for (int i = 0; i < 8; ++i) { r[i] = lo[i]; r[8 + i] = hi[i]; }
        return r;
    };

    v8f acc[8];

    // ================= GEMM1: h = relu(feats @ W1 + b1) =================
    #pragma unroll
    for (int nt = 0; nt < 8; ++nt) {
        float bv = sB1[nt * 16 + l15];
        #pragma unroll
        for (int r = 0; r < 8; ++r) acc[nt][r] = bv;
    }
    #pragma unroll
    for (int ks = 0; ks < 4; ++ks) {
        v16h a = loadA(ks * 32);
        #pragma unroll
        for (int nt = 0; nt < 8; ++nt) {
            v16h b = loadB(sW1, nt, ks * 32);
            acc[nt] = __builtin_amdgcn_wmma_f32_16x16x32_f16(
                false, a, false, b, (short)0, acc[nt], false, false);
        }
    }
    // relu + write hidden back into this wave's private sA rows (reads done;
    // LDS ops are in-order within a wave, rows are wave-private -> no barrier)
    #pragma unroll
    for (int nt = 0; nt < 8; ++nt) {
        #pragma unroll
        for (int r = 0; r < 8; ++r) {
            float v = acc[nt][r];
            v = v > 0.0f ? v : 0.0f;
            int m = wv * 16 + lhalf * 8 + r;
            sA[m * LSTR + nt * 16 + l15] = (_Float16)v;
        }
    }

    // ================= GEMM2: out = h @ W2 + b2 =================
    #pragma unroll
    for (int nt = 0; nt < 8; ++nt) {
        float bv = sB2[nt * 16 + l15];
        #pragma unroll
        for (int r = 0; r < 8; ++r) acc[nt][r] = bv;
    }
    #pragma unroll
    for (int ks = 0; ks < 4; ++ks) {
        v16h a = loadA(ks * 32);
        #pragma unroll
        for (int nt = 0; nt < 8; ++nt) {
            v16h b = loadB(sW2, nt, ks * 32);
            acc[nt] = __builtin_amdgcn_wmma_f32_16x16x32_f16(
                false, a, false, b, (short)0, acc[nt], false, false);
        }
    }

    // ---- streamed (non-temporal) stores: 410 MB written once ----
    #pragma unroll
    for (int nt = 0; nt < 8; ++nt) {
        #pragma unroll
        for (int r = 0; r < 8; ++r) {
            int m = wv * 16 + lhalf * 8 + r;
            int e = e0 + m;
            if (e < nE)
                __builtin_nontemporal_store(acc[nt][r],
                    out + (size_t)e * F_OUT + nt * 16 + l15);
        }
    }
}

// ---------------------------------------------------------------------------
extern "C" void kernel_launch(void* const* d_in, const int* in_sizes, int n_in,
                              void* d_out, int out_size, void* d_ws, size_t ws_size,
                              hipStream_t stream) {
    const float* x    = (const float*)d_in[0];
    const int*   eidx = (const int*)  d_in[1];   // row indices = first E entries
    const float* ea   = (const float*)d_in[2];
    // d_in[3] = u (unused), d_in[4] = batch (unused)
    const float* W1   = (const float*)d_in[5];
    const float* b1   = (const float*)d_in[6];
    const float* W2   = (const float*)d_in[7];
    const float* b2   = (const float*)d_in[8];
    float* out = (float*)d_out;

    const int nE = in_sizes[2] / F_IN;           // edge_attr elements / 64

    _Float16* w1t = (_Float16*)d_ws;             // 32 KB
    _Float16* w2t = w1t + K1 * HIDDEN;           // 32 KB

    prep_weights<<<128, THREADS, 0, stream>>>(W1, W2, w1t, w2t);

    const int nb = (nE + E_PER_BLOCK - 1) / E_PER_BLOCK;
    edge_mlp<<<nb, THREADS, 0, stream>>>(x, eidx, ea, b1, b2, w1t, w2t, out, nE);
}
